// MultiChannelAttentionLayer_45071386805043
// MI455X (gfx1250) — matrix operand order repaired and checked by reading, not compiled, not run on hardware
//
#include <hip/hip_runtime.h>
#include <hip/hip_bf16.h>

// ---------------- problem constants ----------------
#define DIMV   1024
#define HEADS  16
#define HD     64
#define BATCH  2
#define SEQ    2048
#define ROWS   (BATCH*SEQ)        // 4096 flattened tokens
#define NQKV   (3*DIMV)           // 3072
#define SCALEQ 0.03125f           // 1024^-0.5 == 2^-5 (exact)

// ---------------- vector types for WMMA ----------------
typedef __attribute__((ext_vector_type(16))) __bf16         v16bf;
typedef __attribute__((ext_vector_type(8)))  float          v8f;
typedef __attribute__((ext_vector_type(8)))  unsigned short u16x8;
typedef __attribute__((ext_vector_type(16))) unsigned short u16x16;

// f32 -> bf16 bits, round-to-nearest-even
__device__ __forceinline__ unsigned short f32_to_bf16u(float f) {
    unsigned int u = __builtin_bit_cast(unsigned int, f);
    u += 0x7FFFu + ((u >> 16) & 1u);
    return (unsigned short)(u >> 16);
}

// A fragment 16x32 bf16 from row-major [m][k] (ld = elements per row).
// Lanes 0-15: M=lane, K = 0..15; lanes 16-31: M=lane-16, K = 16..31.
__device__ __forceinline__ v16bf load_a_frag(const unsigned short* base, int ld, int lane) {
    const int half = lane >> 4;
    const int m    = lane & 15;
    const unsigned short* p = base + m * ld + half * 16;
    union { u16x16 v; u16x8 h[2]; } u;
    u.h[0] = *(const u16x8*)(p);       // K = half*16 + 0..7
    u.h[1] = *(const u16x8*)(p + 8);   // K = half*16 + 8..15
    return __builtin_bit_cast(v16bf, u.v);
}

// B fragment 32x16 bf16 for C = A * W^T, W row-major [n][k] (base pre-offset to (n0,k0)).
// Lane: N = lane&15, K = (lane>>4)*16 + 0..15 -> one contiguous 32B load.
__device__ __forceinline__ v16bf load_b_frag(const unsigned short* W, int ld, int lane) {
    const unsigned short* p = W + (lane & 15) * ld + (lane >> 4) * 16;
    u16x16 u = *(const u16x16*)p;
    return __builtin_bit_cast(v16bf, u);
}

__device__ __forceinline__ v8f wmma_bf16(v16bf a, v16bf b, v8f c) {
    return __builtin_amdgcn_wmma_f32_16x16x32_bf16(false, a, false, b, (short)0, c, false, false);
}

// ---------------- f32 -> bf16 conversion ----------------
__global__ void __launch_bounds__(256) convert_bf16_kernel(const float* __restrict__ in,
                                                           unsigned short* __restrict__ out, int n) {
    int i = blockIdx.x * blockDim.x + threadIdx.x;
    int stride = gridDim.x * blockDim.x;
    for (; i < n; i += stride) out[i] = f32_to_bf16u(in[i]);
}

// ================= shared 32x64-tile GEMM core (C = A @ W^T) =================
// Double-buffered over k (step 32, unrolled by 64). acc[mi][j] covers rows
// m0+mi*16.., cols n0+j*16... A row-major [M][K], W row-major [N][K].
struct Tile32x64 {
    v8f acc[2][4];
};

__device__ __forceinline__ void gemm_tile_32x64(const unsigned short* __restrict__ A,
                                                const unsigned short* __restrict__ W,
                                                int K, int m0, int n0, int lane,
                                                Tile32x64& t) {
#pragma unroll
    for (int mi = 0; mi < 2; ++mi)
#pragma unroll
        for (int j = 0; j < 4; ++j) t.acc[mi][j] = (v8f){};

    const unsigned short* Arow0 = A + (long)m0 * K;
    const unsigned short* Arow1 = A + (long)(m0 + 16) * K;

    v16bf a0[2], b0[4], a1[2], b1[4];
    // prologue: k = 0 into buffer 0
    a0[0] = load_a_frag(Arow0, K, lane);
    a0[1] = load_a_frag(Arow1, K, lane);
#pragma unroll
    for (int j = 0; j < 4; ++j) b0[j] = load_b_frag(W + (long)(n0 + 16 * j) * K, K, lane);

    for (int k0 = 0; k0 < K; k0 += 64) {
        // prefetch k0+32 into buffer 1 (always in range: K multiple of 64)
        a1[0] = load_a_frag(Arow0 + k0 + 32, K, lane);
        a1[1] = load_a_frag(Arow1 + k0 + 32, K, lane);
#pragma unroll
        for (int j = 0; j < 4; ++j)
            b1[j] = load_b_frag(W + (long)(n0 + 16 * j) * K + k0 + 32, K, lane);
        // compute with buffer 0 (k0)
#pragma unroll
        for (int j = 0; j < 4; ++j) {
            t.acc[0][j] = wmma_bf16(a0[0], b0[j], t.acc[0][j]);
            t.acc[1][j] = wmma_bf16(a0[1], b0[j], t.acc[1][j]);
        }
        // prefetch k0+64 into buffer 0 (guarded, scalar-uniform branch)
        if (k0 + 64 < K) {
            a0[0] = load_a_frag(Arow0 + k0 + 64, K, lane);
            a0[1] = load_a_frag(Arow1 + k0 + 64, K, lane);
#pragma unroll
            for (int j = 0; j < 4; ++j)
                b0[j] = load_b_frag(W + (long)(n0 + 16 * j) * K + k0 + 64, K, lane);
        }
        // compute with buffer 1 (k0+32)
#pragma unroll
        for (int j = 0; j < 4; ++j) {
            t.acc[0][j] = wmma_bf16(a1[0], b1[j], t.acc[0][j]);
            t.acc[1][j] = wmma_bf16(a1[1], b1[j], t.acc[1][j]);
        }
    }
}

// ---------------- QKV projection: qkv = x @ w_qkv^T, scatter to heads ----------------
// Q -> [b*h][n][64] (pre-scaled by 1/32), K -> [b*h][n][64], V -> transposed [b*h][64][n]
__global__ void __launch_bounds__(256) gemm_qkv_kernel(const unsigned short* __restrict__ Xb,
                                                       const unsigned short* __restrict__ Wb,
                                                       unsigned short* __restrict__ Qh,
                                                       unsigned short* __restrict__ Kh,
                                                       unsigned short* __restrict__ Vt) {
    const int lane = threadIdx.x & 31;
    const int wid  = blockIdx.x * 8 + (threadIdx.x >> 5);
    const int TN   = NQKV / 64;              // 48 column tiles
    const int m0   = (wid / TN) * 32;
    const int n0   = (wid % TN) * 64;

    Tile32x64 t;
    gemm_tile_32x64(Xb, Wb, DIMV, m0, n0, lane, t);

    const int half = lane >> 4, nl = lane & 15;
#pragma unroll
    for (int j = 0; j < 4; ++j) {
        const int c     = n0 + 16 * j + nl;     // qkv column 0..3071
        const int which = c >> 10;              // 0=q 1=k 2=v
        const int d1    = c & 1023;
        const int h     = d1 >> 6;
        const int d     = d1 & 63;
#pragma unroll
        for (int mi = 0; mi < 2; ++mi) {
#pragma unroll
            for (int r = 0; r < 8; ++r) {
                const int row = m0 + mi * 16 + r + half * 8;  // token index 0..4095
                const int bb  = row >> 11;
                const int n   = row & 2047;
                const int bh  = bb * HEADS + h;
                const float v = t.acc[mi][j][r];
                if (which == 0)      Qh[(bh * SEQ + n) * HD + d] = f32_to_bf16u(v * SCALEQ);
                else if (which == 1) Kh[(bh * SEQ + n) * HD + d] = f32_to_bf16u(v);
                else                 Vt[(bh * HD + d) * SEQ + n] = f32_to_bf16u(v);
            }
        }
    }
}

// ---------------- flash attention: one wave per (b,h, 16-query tile) ----------------
__global__ void __launch_bounds__(256) attn_kernel(const unsigned short* __restrict__ Qh,
                                                   const unsigned short* __restrict__ Kh,
                                                   const unsigned short* __restrict__ Vt,
                                                   unsigned short* __restrict__ Ob) {
    __shared__ unsigned short lds_p[8][16 * 32];   // 1KB P-tile bounce per wave
    const int lane  = threadIdx.x & 31;
    const int wslot = threadIdx.x >> 5;
    const int wid   = blockIdx.x * 8 + wslot;
    const int QT    = SEQ / 16;                    // 128 query tiles per (b,h)
    const int bh    = wid / QT;
    const int qt    = wid % QT;
    const int half  = lane >> 4, nl = lane & 15;

    const unsigned short* Q = Qh + (long)bh * SEQ * HD + qt * 16 * HD;
    const unsigned short* K = Kh + (long)bh * SEQ * HD;
    const unsigned short* V = Vt + (long)bh * HD * SEQ;

    const v16bf q0 = load_a_frag(Q + 0,  HD, lane);   // d = 0..31
    const v16bf q1 = load_a_frag(Q + 32, HD, lane);   // d = 32..63

    v8f o[4] = {};
    float mrow[8], lrow[8];
#pragma unroll
    for (int r = 0; r < 8; ++r) { mrow[r] = -3.0e38f; lrow[r] = 0.0f; }

    unsigned short* pl = lds_p[wslot];

    for (int kk = 0; kk < SEQ; kk += 32) {
        // S = (Q * 1/32) @ K^T  -> two 16x16 fragments (keys kk..kk+15, kk+16..kk+31)
        v8f s0 = {}, s1 = {};
        s0 = wmma_bf16(q0, load_b_frag(K + (kk     ) * HD + 0,  HD, lane), s0);
        s0 = wmma_bf16(q1, load_b_frag(K + (kk     ) * HD + 32, HD, lane), s0);
        s1 = wmma_bf16(q0, load_b_frag(K + (kk + 16) * HD + 0,  HD, lane), s1);
        s1 = wmma_bf16(q1, load_b_frag(K + (kk + 16) * HD + 32, HD, lane), s1);

        // prefetch V fragments now -> VMEM latency hides under the softmax VALU work
        v16bf vb[4];
#pragma unroll
        for (int j = 0; j < 4; ++j)
            vb[j] = load_b_frag(V + (j * 16) * SEQ + kk, SEQ, lane);

        // online softmax; each lane owns rows {r + half*8} (same mapping in every fragment)
#pragma unroll
        for (int r = 0; r < 8; ++r) {
            float cand = fmaxf(s0[r], s1[r]);
#pragma unroll
            for (int off = 1; off < 16; off <<= 1)
                cand = fmaxf(cand, __shfl_xor(cand, off, 32));
            const float mnew  = fmaxf(mrow[r], cand);
            const float alpha = exp2f((mrow[r] - mnew) * 1.44269504f);
            const float p0    = exp2f((s0[r] - mnew) * 1.44269504f);
            const float p1    = exp2f((s1[r] - mnew) * 1.44269504f);
            float rs = p0 + p1;
#pragma unroll
            for (int off = 1; off < 16; off <<= 1)
                rs += __shfl_xor(rs, off, 32);
            lrow[r] = lrow[r] * alpha + rs;
            mrow[r] = mnew;
#pragma unroll
            for (int j = 0; j < 4; ++j) o[j][r] *= alpha;
            // stash P (bf16) row-major [16][32] in LDS for D-frag -> A-frag re-layout
            pl[(r + half * 8) * 32 + nl]      = f32_to_bf16u(p0);
            pl[(r + half * 8) * 32 + 16 + nl] = f32_to_bf16u(p1);
        }

        // O += P @ V   (A = P 16x32 from LDS, B = prefetched V fragments)
        const v16bf pa = load_a_frag(pl, 32, lane);
#pragma unroll
        for (int j = 0; j < 4; ++j)
            o[j] = wmma_bf16(pa, vb[j], o[j]);
    }

    // normalize and store O as bf16 row-major [4096][1024] for the output GEMM
    const int qbase = (bh / HEADS) * SEQ + qt * 16;  // global token row base
    const int h     = bh % HEADS;
#pragma unroll
    for (int j = 0; j < 4; ++j) {
#pragma unroll
        for (int r = 0; r < 8; ++r) {
            const int row = qbase + r + half * 8;
            Ob[(long)row * DIMV + h * HD + j * 16 + nl] = f32_to_bf16u(o[j][r] / lrow[r]);
        }
    }
}

// ---------------- output projection: out = O @ w_out^T + b_out (f32) ----------------
__global__ void __launch_bounds__(256) gemm_out_kernel(const unsigned short* __restrict__ Ob,
                                                       const unsigned short* __restrict__ Wb,
                                                       const float* __restrict__ bias,
                                                       float* __restrict__ out) {
    const int lane = threadIdx.x & 31;
    const int wid  = blockIdx.x * 8 + (threadIdx.x >> 5);
    const int TN   = DIMV / 64;              // 16 column tiles
    const int m0   = (wid / TN) * 32;
    const int n0   = (wid % TN) * 64;

    Tile32x64 t;
    gemm_tile_32x64(Ob, Wb, DIMV, m0, n0, lane, t);

    const int half = lane >> 4, nl = lane & 15;
#pragma unroll
    for (int j = 0; j < 4; ++j) {
        const int c  = n0 + 16 * j + nl;
        const float bv = bias[c];
#pragma unroll
        for (int mi = 0; mi < 2; ++mi) {
#pragma unroll
            for (int r = 0; r < 8; ++r) {
                const int row = m0 + mi * 16 + r + half * 8;
                out[(long)row * DIMV + c] = t.acc[mi][j][r] + bv;
            }
        }
    }
}

// ---------------- launch ----------------
extern "C" void kernel_launch(void* const* d_in, const int* in_sizes, int n_in,
                              void* d_out, int out_size, void* d_ws, size_t ws_size,
                              hipStream_t stream) {
    const float* x     = (const float*)d_in[0];
    const float* w_qkv = (const float*)d_in[1];
    const float* w_out = (const float*)d_in[2];
    const float* b_out = (const float*)d_in[3];
    float* out = (float*)d_out;

    // workspace layout (bf16 bits), ~48 MB total
    unsigned short* Xb  = (unsigned short*)d_ws;            // 4096*1024
    unsigned short* Wqb = Xb  + (size_t)ROWS * DIMV;        // 3072*1024
    unsigned short* Wob = Wqb + (size_t)NQKV * DIMV;        // 1024*1024
    unsigned short* Qh  = Wob + (size_t)DIMV * DIMV;        // 32*2048*64
    unsigned short* Kh  = Qh  + (size_t)ROWS * DIMV;
    unsigned short* Vt  = Kh  + (size_t)ROWS * DIMV;
    unsigned short* Ob  = Vt  + (size_t)ROWS * DIMV;

    convert_bf16_kernel<<<512, 256, 0, stream>>>(x,     Xb,  ROWS * DIMV);
    convert_bf16_kernel<<<512, 256, 0, stream>>>(w_qkv, Wqb, NQKV * DIMV);
    convert_bf16_kernel<<<256, 256, 0, stream>>>(w_out, Wob, DIMV * DIMV);

    gemm_qkv_kernel<<<(ROWS / 32) * (NQKV / 64) / 8, 256, 0, stream>>>(Xb, Wqb, Qh, Kh, Vt);
    attn_kernel<<<(BATCH * HEADS * (SEQ / 16)) / 8, 256, 0, stream>>>(Qh, Kh, Vt, Ob);
    gemm_out_kernel<<<(ROWS / 32) * (DIMV / 64) / 8, 256, 0, stream>>>(Ob, Wob, b_out, out);
}
